// DAReLoss_60730837566259
// MI455X (gfx1250) — matrix operand order repaired and checked by reading, not compiled, not run on hardware
//
#include <hip/hip_runtime.h>
#include <hip/hip_bf16.h>

// ---------------------------------------------------------------------------
// DAReLoss for MI455X (gfx1250, wave32, WMMA).
// B=4096, D=512, N=2B=8192. TAU=1, LAM=1.
// Dominant cost: sim = Xn @ Xn^T (8192x8192x512, 68.7 GFLOP) -> done with
// v_wmma_f32_16x16x32_bf16 on an L2-resident 8MB bf16 operand, fused with an
// online (flash-style) row logsumexp. Each wave processes 2 column tiles per
// iteration (one A fragment feeds 2 WMMAs -> 3 ds_load_b128 per WMMA).
// B tiles stream memory->LDS via the gfx1250 async-to-LDS path (ASYNCcnt)
// when the builtin is available.
// ---------------------------------------------------------------------------

#ifndef __has_builtin
#define __has_builtin(x) 0
#endif
#if __has_builtin(__builtin_amdgcn_global_load_async_to_lds_b128)
#define ASYNC_LDS 1
#else
#define ASYNC_LDS 0
#endif

#define Bsz 4096
#define Dd  512
#define Nn  8192
#define SPLITS 8
#define JT_PER_SPLIT ((Nn / 16) / SPLITS)  // 64 column tiles per split
#define NPAIR (JT_PER_SPLIT / 2)           // 32 pair iterations
#define MBLK 64                            // rows per workgroup (4 waves x 16)

typedef __attribute__((ext_vector_type(16))) __bf16   v16bf;
typedef __attribute__((ext_vector_type(8)))  float    v8f;
typedef __attribute__((ext_vector_type(4)))  unsigned u32x4;
typedef __attribute__((ext_vector_type(4)))  int      i32x4;

union Frag { v16bf v; u32x4 u[2]; };

__device__ __forceinline__ unsigned short f2bf(float f) {
  unsigned u = __float_as_uint(f);
  u += 0x7FFFu + ((u >> 16) & 1u);     // round-to-nearest-even
  return (unsigned short)(u >> 16);
}
__device__ __forceinline__ float bf2f(unsigned short h) {
  return __uint_as_float(((unsigned)h) << 16);
}
// Counter-based deterministic PRNG (PCG hash) for the fixed-noise reparam.
__device__ __forceinline__ unsigned pcg(unsigned x) {
  x = x * 747796405u + 2891336453u;
  unsigned w = ((x >> ((x >> 28) + 4u)) ^ x) * 277803737u;
  return (w >> 22) ^ w;
}

__device__ __forceinline__ float blkred256(float* sd, float v, int tid) {
  sd[tid] = v;
  __syncthreads();
  for (int s = 128; s > 0; s >>= 1) {
    if (tid < s) sd[tid] += sd[tid + s];
    __syncthreads();
  }
  float r = sd[0];
  __syncthreads();
  return r;
}

#if ASYNC_LDS
typedef __attribute__((address_space(1))) i32x4* gas_p;  // global int4*
typedef __attribute__((address_space(3))) i32x4* las_p;  // LDS int4*
// Copy one 32-row x 512-col bf16 tile pair (32 KB) memory -> LDS, async.
// 128 threads x 16 x b128. Tracked by ASYNCcnt.
__device__ __forceinline__ void copy_pair_async(unsigned short* dst,
                                                const unsigned short* src,
                                                int tid) {
#pragma unroll
  for (int i = 0; i < 16; ++i) {
    const int off = (tid + i * 128) * 8;  // 8 shorts = 16 bytes
    __builtin_amdgcn_global_load_async_to_lds_b128(
        (gas_p)(src + off), (las_p)(dst + off), 0, 0);
  }
}
__device__ __forceinline__ void wait_async0() {
#if __has_builtin(__builtin_amdgcn_s_wait_asynccnt)
  __builtin_amdgcn_s_wait_asynccnt(0);
#else
  asm volatile("s_wait_asynccnt 0" ::: "memory");
#endif
}
#endif

// ---------------------------------------------------------------------------
// 1) Reparameterize x = mu + exp(0.5*logvar)*eps, L2-normalize rows, store
//    bf16 features Xn [N x D]. One block per row.
// ---------------------------------------------------------------------------
__global__ __launch_bounds__(256) void k_reparam(
    const float* __restrict__ mu,  const float* __restrict__ muc,
    const float* __restrict__ lv,  const float* __restrict__ lvc,
    unsigned short* __restrict__ Xn) {
  __shared__ float sd[256];
  const int row = blockIdx.x, tid = threadIdx.x;
  const float* m = (row < Bsz) ? (mu + (size_t)row * Dd) : (muc + (size_t)(row - Bsz) * Dd);
  const float* l = (row < Bsz) ? (lv + (size_t)row * Dd) : (lvc + (size_t)(row - Bsz) * Dd);
  float x[2];
  float ss = 0.f;
#pragma unroll
  for (int q = 0; q < 2; ++q) {
    const int d = tid + q * 256;
    const unsigned idx = (unsigned)(row * Dd + d);
    const unsigned h1 = pcg(idx * 2u + 0x9E3779B9u);
    const unsigned h2 = pcg(idx * 2u + 0x85EBCA6Bu);
    const float u1 = (float)(h1 >> 8) * (1.0f / 16777216.0f) + 5.96e-8f;  // (0,1]
    const float u2 = (float)(h2 >> 8) * (1.0f / 16777216.0f);
    const float eps = sqrtf(-2.0f * __logf(u1)) * __cosf(6.28318530718f * u2);
    x[q] = m[d] + __expf(0.5f * l[d]) * eps;
    ss += x[q] * x[q];
  }
  const float tot = blkred256(sd, ss, tid);
  const float inv = 1.0f / fmaxf(sqrtf(tot), 1e-12f);
#pragma unroll
  for (int q = 0; q < 2; ++q)
    Xn[(size_t)row * Dd + tid + q * 256] = f2bf(x[q] * inv);
}

// ---------------------------------------------------------------------------
// 2) WMMA GEMM strip + fused online logsumexp.
//    Grid: (Nn/MBLK, SPLITS). Block: 128 threads = 4 waves, each wave owns 16
//    rows and processes 2 column tiles per iteration. A strip (64x512 bf16)
//    in LDS; paired B tiles (32x512 bf16) double-buffered via async-to-LDS.
// ---------------------------------------------------------------------------
__global__ __launch_bounds__(128) void k_gemm_lse(
    const unsigned short* __restrict__ Xn,
    float* __restrict__ pm, float* __restrict__ ps) {
  extern __shared__ char smem[];
  unsigned short* sA = (unsigned short*)smem;                    // 64 x 512 bf16
  unsigned short* sB = (unsigned short*)(smem + MBLK * Dd * 2);  // 2 x 32 x 512

  const int tid  = threadIdx.x;
  const int lane = tid & 31;
  const int wv   = tid >> 5;
  const int col  = lane & 15;
  const int half = lane >> 4;
  const int rowBlock = blockIdx.x * MBLK;
  const int split    = blockIdx.y;

  {  // A strip: 64 rows, contiguous in global -> LDS (32 x u32x4 per thread)
    const u32x4* g = (const u32x4*)(Xn + (size_t)rowBlock * Dd);
    u32x4* s = (u32x4*)sA;
#pragma unroll
    for (int i = 0; i < (MBLK * Dd / 8) / 128; ++i) s[tid + i * 128] = g[tid + i * 128];
  }
  // Prologue: first B tile pair of this split.
  {
    const unsigned short* src0 = Xn + (size_t)(split * JT_PER_SPLIT) * 16 * Dd;
#if ASYNC_LDS
    copy_pair_async(sB, src0, tid);
    wait_async0();
#else
    const u32x4* g = (const u32x4*)src0;
    u32x4* s = (u32x4*)sB;
#pragma unroll
    for (int i = 0; i < 16; ++i) s[tid + i * 128] = g[tid + i * 128];
#endif
  }
  __syncthreads();

  float M[8], S[8];
#pragma unroll
  for (int r = 0; r < 8; ++r) { M[r] = -__builtin_inff(); S[r] = 0.f; }

  // A fragment (16x32 bf16): lane holds row (lane&15), 16 contiguous K
  // starting at half*16. Two 16B LDS loads per k-step.
  const unsigned short* aRow = sA + (size_t)(wv * 16 + col) * Dd;
  const int dA = half * 16;
  // B fragment (32x16 bf16): lane holds column (lane&15), 16 contiguous K.
  const int dB = half * 16;

  for (int p = 0; p < NPAIR; ++p) {
    const int basetile = split * JT_PER_SPLIT + 2 * p;
    const bool haveNext = (p + 1) < NPAIR;

    // Kick off next pair's memory->LDS transfer (overlaps with WMMAs).
    unsigned short* nbuf = sB + (size_t)((p + 1) & 1) * 32 * Dd;
    const unsigned short* nsrc = Xn + (size_t)(basetile + 2) * 16 * Dd;
#if ASYNC_LDS
    if (haveNext) copy_pair_async(nbuf, nsrc, tid);
#else
    u32x4 pre[16];
    if (haveNext) {
      const u32x4* g = (const u32x4*)nsrc;
#pragma unroll
      for (int i = 0; i < 16; ++i) pre[i] = g[tid + i * 128];
    }
#endif
    if (p + 2 < NPAIR)  // warm L2 for the pair after that
      __builtin_prefetch(Xn + (size_t)(basetile + 4) * 16 * Dd + tid * 16, 0, 0);

    const unsigned short* bBuf  = sB + (size_t)(p & 1) * 32 * Dd;
    const unsigned short* bRow0 = bBuf + (size_t)col * Dd;
    const unsigned short* bRow1 = bBuf + (size_t)(16 + col) * Dd;

    v8f acc0 = {0.f, 0.f, 0.f, 0.f, 0.f, 0.f, 0.f, 0.f};
    v8f acc1 = {0.f, 0.f, 0.f, 0.f, 0.f, 0.f, 0.f, 0.f};
#pragma unroll
    for (int k0 = 0; k0 < Dd; k0 += 32) {
      Frag a, b0, b1;
      a.u[0]  = *(const u32x4*)(aRow + k0 + dA);
      a.u[1]  = *(const u32x4*)(aRow + k0 + dA + 8);
      b0.u[0] = *(const u32x4*)(bRow0 + k0 + dB);
      b0.u[1] = *(const u32x4*)(bRow0 + k0 + dB + 8);
      b1.u[0] = *(const u32x4*)(bRow1 + k0 + dB);
      b1.u[1] = *(const u32x4*)(bRow1 + k0 + dB + 8);
      acc0 = __builtin_amdgcn_wmma_f32_16x16x32_bf16(
          false, a.v, false, b0.v, (short)0, acc0, false, false);
      acc1 = __builtin_amdgcn_wmma_f32_16x16x32_bf16(
          false, a.v, false, b1.v, (short)0, acc1, false, false);
    }

#if !ASYNC_LDS
    if (haveNext) {
      u32x4* s = (u32x4*)nbuf;
#pragma unroll
      for (int i = 0; i < 16; ++i) s[tid + i * 128] = pre[i];
    }
#endif

    // Online softmax update over both tiles. C layout: VGPR r -> row r
    // (lanes 0-15) or 8+r (lanes 16-31), column = lane&15. TAU==1, no scale.
    const int gc0 = basetile * 16 + col;
    const int gc1 = gc0 + 16;
    float* a0 = (float*)&acc0;
    float* a1 = (float*)&acc1;
#pragma unroll
    for (int r = 0; r < 8; ++r) {
      const int grow = rowBlock + wv * 16 + r + 8 * half;
      float v0 = a0[r], v1 = a1[r];
      if (gc0 == grow) v0 = -__builtin_inff();  // mask diagonal
      if (gc1 == grow) v1 = -__builtin_inff();
      float tm = fmaxf(v0, v1);
#pragma unroll
      for (int m = 1; m < 16; m <<= 1) tm = fmaxf(tm, __shfl_xor(tm, m, 32));
      const float nm = fmaxf(M[r], tm);
      float e = __expf(v0 - nm) + __expf(v1 - nm);
#pragma unroll
      for (int m = 1; m < 16; m <<= 1) e += __shfl_xor(e, m, 32);
      S[r] = S[r] * __expf(M[r] - nm) + e;
      M[r] = nm;
    }

#if ASYNC_LDS
    if (haveNext) wait_async0();
#endif
    __syncthreads();
  }

  if (col == 0) {  // lanes 0 and 16 publish 8 rows each
#pragma unroll
    for (int r = 0; r < 8; ++r) {
      const int grow = rowBlock + wv * 16 + r + 8 * half;
      pm[(size_t)grow * SPLITS + split] = M[r];
      ps[(size_t)grow * SPLITS + split] = S[r];
    }
  }
}

// ---------------------------------------------------------------------------
// 3) Merge SPLITS partial (max, sum) pairs per row into lse[i].
// ---------------------------------------------------------------------------
__global__ __launch_bounds__(256) void k_combine(
    const float* __restrict__ pm, const float* __restrict__ ps,
    float* __restrict__ lse) {
  const int i = blockIdx.x * 256 + threadIdx.x;
  if (i >= Nn) return;
  float mm = -__builtin_inff();
#pragma unroll
  for (int s = 0; s < SPLITS; ++s) mm = fmaxf(mm, pm[(size_t)i * SPLITS + s]);
  float sum = 0.f;
#pragma unroll
  for (int s = 0; s < SPLITS; ++s)
    sum += ps[(size_t)i * SPLITS + s] * __expf(pm[(size_t)i * SPLITS + s] - mm);
  lse[i] = mm + __logf(sum);
}

// ---------------------------------------------------------------------------
// 4) SupCon per-row term: exactly one positive, partner = (i+B) mod N.
//    termA[i] = dot(Xn_i, Xn_partner)/TAU - lse[i].
// ---------------------------------------------------------------------------
__global__ __launch_bounds__(128) void k_pos(
    const unsigned short* __restrict__ Xn, const float* __restrict__ lse,
    float* __restrict__ termA) {
  const int i = blockIdx.x;
  const int p = (i + Bsz) & (Nn - 1);
  const int tid = threadIdx.x;
  const unsigned short* a = Xn + (size_t)i * Dd;
  const unsigned short* b = Xn + (size_t)p * Dd;
  float s = 0.f;
#pragma unroll
  for (int q = 0; q < 4; ++q) {
    const int d = tid + q * 128;
    s += bf2f(a[d]) * bf2f(b[d]);
  }
  for (int m = 1; m < 32; m <<= 1) s += __shfl_xor(s, m, 32);
  __shared__ float sw[4];
  if ((tid & 31) == 0) sw[tid >> 5] = s;
  __syncthreads();
  if (tid == 0) termA[i] = (sw[0] + sw[1] + sw[2] + sw[3]) - lse[i];  // TAU=1
}

// ---------------------------------------------------------------------------
// 5) JSD between Gaussians on L2-normalized stats. One block per row of B.
//    termJ[b] = 0.5*(kl1+kl2) = 0.25 * row-sum of per-element integrands.
// ---------------------------------------------------------------------------
__global__ __launch_bounds__(256) void k_jsd(
    const float* __restrict__ mu,  const float* __restrict__ muc,
    const float* __restrict__ lv,  const float* __restrict__ lvc,
    float* __restrict__ termJ) {
  __shared__ float sd[256];
  const int b = blockIdx.x, tid = threadIdx.x;
  const float* pmu = mu + (size_t)b * Dd;
  const float* pmc = muc + (size_t)b * Dd;
  const float* plv = lv + (size_t)b * Dd;
  const float* plc = lvc + (size_t)b * Dd;
  float am[2], ac[2], al[2], aq[2];
  float s0 = 0.f, s1 = 0.f, s2 = 0.f, s3 = 0.f;
#pragma unroll
  for (int q = 0; q < 2; ++q) {
    const int d = tid + q * 256;
    am[q] = pmu[d]; ac[q] = pmc[d]; al[q] = plv[d]; aq[q] = plc[d];
    s0 += am[q] * am[q]; s1 += ac[q] * ac[q];
    s2 += al[q] * al[q]; s3 += aq[q] * aq[q];
  }
  const float i0 = 1.f / fmaxf(sqrtf(blkred256(sd, s0, tid)), 1e-12f);
  const float i1 = 1.f / fmaxf(sqrtf(blkred256(sd, s1, tid)), 1e-12f);
  const float i2 = 1.f / fmaxf(sqrtf(blkred256(sd, s2, tid)), 1e-12f);
  const float i3 = 1.f / fmaxf(sqrtf(blkred256(sd, s3, tid)), 1e-12f);
  float t = 0.f;
#pragma unroll
  for (int q = 0; q < 2; ++q) {
    const float mn = am[q] * i0, mc_ = ac[q] * i1;
    const float ln = al[q] * i2, lc_ = aq[q] * i3;
    const float mm = 0.5f * (mn + mc_);
    const float v1 = __expf(ln), v2 = __expf(lc_);
    const float vm = 0.5f * (v1 + v2) + 1e-8f;   // exp(log_var_m)
    const float lvm = __logf(vm);
    const float den = vm + 1e-8f;
    const float d1 = mn - mm, d2 = mc_ - mm;
    t += (v1 + 1e-8f) / den + (lvm - ln) + d1 * d1 / den - 1.f;
    t += (v2 + 1e-8f) / den + (lvm - lc_) + d2 * d2 / den - 1.f;
  }
  const float tot = blkred256(sd, t, tid);
  if (tid == 0) termJ[b] = 0.25f * tot;
}

// ---------------------------------------------------------------------------
// 6) Deterministic final reduction (fixed per-thread order + fixed tree).
// ---------------------------------------------------------------------------
__global__ __launch_bounds__(256) void k_final(
    const float* __restrict__ termA, const float* __restrict__ termJ,
    float* __restrict__ out) {
  __shared__ float sd[256];
  const int tid = threadIdx.x;
  float a = 0.f, j = 0.f;
  for (int i = tid; i < Nn; i += 256) a += termA[i];
  for (int i = tid; i < Bsz; i += 256) j += termJ[i];
  const float v = a * (-1.0f / ((float)Nn * (1.0f + 1e-5f)))  // SupCon
                + j * (1.0f / (float)Bsz);                     // LAM * mean JSD
  const float tot = blkred256(sd, v, tid);
  if (tid == 0) out[0] = tot;
}

// ---------------------------------------------------------------------------
extern "C" void kernel_launch(void* const* d_in, const int* in_sizes, int n_in,
                              void* d_out, int out_size, void* d_ws, size_t ws_size,
                              hipStream_t stream) {
  const float* mu  = (const float*)d_in[0];
  const float* muc = (const float*)d_in[1];
  const float* lv  = (const float*)d_in[2];
  const float* lvc = (const float*)d_in[3];

  // Workspace layout (~9 MB): bf16 features, softmax partials, lse, terms.
  char* ws = (char*)d_ws;
  unsigned short* Xn = (unsigned short*)ws;                  // N*D bf16 = 8 MB
  float* pm    = (float*)(ws + (size_t)Nn * Dd * 2);         // N*SPLITS
  float* ps    = pm + (size_t)Nn * SPLITS;                   // N*SPLITS
  float* lse   = ps + (size_t)Nn * SPLITS;                   // N
  float* termA = lse + Nn;                                   // N
  float* termJ = termA + Nn;                                 // B

  const size_t lds_gemm = (size_t)(MBLK * Dd + 2 * 32 * Dd) * 2;  // 128 KB

  k_reparam<<<Nn, 256, 0, stream>>>(mu, muc, lv, lvc, Xn);
  k_gemm_lse<<<dim3(Nn / MBLK, SPLITS), 128, lds_gemm, stream>>>(Xn, pm, ps);
  k_combine<<<(Nn + 255) / 256, 256, 0, stream>>>(pm, ps, lse);
  k_pos<<<Nn, 128, 0, stream>>>(Xn, lse, termA);
  k_jsd<<<Bsz, 256, 0, stream>>>(mu, muc, lv, lvc, termJ);
  k_final<<<1, 256, 0, stream>>>(termA, termJ, (float*)d_out);
}
